// MistralFlashAttention2_62010737819750
// MI455X (gfx1250) — compile-verified, matrix operand order checked by reading, not run on hardware
//
#include <hip/hip_runtime.h>
#include <hip/hip_bf16.h>

typedef __attribute__((ext_vector_type(16))) _Float16 v16h;
typedef __attribute__((ext_vector_type(8)))  _Float16 v8h;
typedef __attribute__((ext_vector_type(8)))  float    v8f;
typedef __attribute__((ext_vector_type(4)))  unsigned int v4u;
typedef __attribute__((ext_vector_type(8)))  int      v8i;
typedef __attribute__((ext_vector_type(4)))  int      v4i;

#define HIDDEN   4096
#define NH       32
#define NKV      8
#define HD       128
#define SEQ      2048
#define BATCH    2
#define NROWS    (BATCH*SEQ)          // 4096
#define SCALE    0.08838834764831845f // 1/sqrt(128)

#if __has_builtin(__builtin_amdgcn_tensor_load_to_lds)
#define HAVE_TDM 1
#else
#define HAVE_TDM 0
#endif

// CDNA5 async global->LDS copy of 16 bytes (ASYNCcnt-tracked).
__device__ __forceinline__ void async_cp16(unsigned lds_off, const void* g) {
    asm volatile("global_load_async_to_lds_b128 %0, %1, off"
                 :: "v"(lds_off), "v"((unsigned long long)g) : "memory");
}
__device__ __forceinline__ void wait_async0() {
    asm volatile("s_wait_asynccnt 0x0" ::: "memory");
}
__device__ __forceinline__ unsigned lds_off32(const void* p) {
    return (unsigned)(unsigned long long)p;   // low 32 bits of LDS aperture addr
}

// TDM: DMA a contiguous 8KB (4096 x 2B) tile from global to LDS.
// Descriptor per ISA ch.8: group0 = {flags,count=1 | lds_addr | gaddr | type=2},
// group1 = {data_size=2B, tensor_dim0=256Ki, tensor_dim1=1, tile_dim0=4096,
//           tile_dim1=1, dim0_stride=256Ki}.
__device__ __forceinline__ void tdm_copy_8k(unsigned lds_addr, unsigned long long ga) {
#if HAVE_TDM
    v4u g0;
    g0[0] = 1u;                                   // count=1, user mode
    g0[1] = lds_addr;                             // lds_addr (bytes)
    g0[2] = (unsigned)ga;                         // global_addr[31:0]
    g0[3] = (unsigned)((ga >> 32) & 0x1FFFFFFu) | (2u << 30);  // addr[56:32] | type=2
    v8i g1;
    g1[0] = (1 << 16);                            // wg_mask=0, data_size=1 (2B)
    g1[1] = 0;                                    // tensor_dim0[15:0] (=0x40000 lo16)
    g1[2] = 4 | (1 << 16);                        // tensor_dim0 hi=4, tensor_dim1=1
    g1[3] = (4096 << 16);                         // tensor_dim1 hi=0, tile_dim0=4096
    g1[4] = 1;                                    // tile_dim1=1, tile_dim2=0
    g1[5] = 262144;                               // tensor_dim0_stride lo32
    g1[6] = 0;
    g1[7] = 0;
    v4i z4 = {0, 0, 0, 0};
#if __clang_major__ >= 23
    v8i z8 = {0, 0, 0, 0, 0, 0, 0, 0};
    __builtin_amdgcn_tensor_load_to_lds(g0, g1, z4, z4, z8, 0);
#else
    __builtin_amdgcn_tensor_load_to_lds(g0, g1, z4, z4, 0);
#endif
#endif
}

// ---------------------------------------------------------------------------
// vectorized fp32 -> f16 (8 elems/thread)
__global__ void k_f32_to_f16(const float* __restrict__ src, _Float16* __restrict__ dst, int n8) {
    int i = blockIdx.x * blockDim.x + threadIdx.x;
    if (i >= n8) return;
    const float4* s = (const float4*)src;
    float4 a = s[2 * i], b = s[2 * i + 1];
    v8h o;
    o[0] = (_Float16)a.x; o[1] = (_Float16)a.y; o[2] = (_Float16)a.z; o[3] = (_Float16)a.w;
    o[4] = (_Float16)b.x; o[5] = (_Float16)b.y; o[6] = (_Float16)b.z; o[7] = (_Float16)b.w;
    *(v8h*)(dst + 8 * i) = o;
}

// LDS-tiled transpose + convert: W[K][N] fp32 -> Wt[N][K] f16 (coalesced both ways)
__global__ void __launch_bounds__(256) k_transp_f16(const float* __restrict__ W,
                                                    _Float16* __restrict__ Wt, int K, int N) {
    __shared__ float tile[32][33];
    int ntiles = N >> 5;
    int bx = blockIdx.x % ntiles;       // n-tile
    int by = blockIdx.x / ntiles;       // k-tile
    int tx = threadIdx.x & 31;
    int ty = threadIdx.x >> 5;          // 0..7
#pragma unroll
    for (int i = 0; i < 4; ++i) {
        int k = by * 32 + ty + i * 8;
        tile[ty + i * 8][tx] = W[(size_t)k * N + bx * 32 + tx];
    }
    __syncthreads();
#pragma unroll
    for (int i = 0; i < 4; ++i) {
        int n = bx * 32 + ty + i * 8;
        Wt[(size_t)n * K + by * 32 + tx] = (_Float16)tile[tx][ty + i * 8];
    }
}

// ---------------------------------------------------------------------------
// GEMM: C[M][N] f32 = A[M][K] f16 (row major) * Bt[N][K] f16 (B transposed).
// One wave computes a 32x64 tile: 2 A-frags x 4 B-frags -> 8 WMMAs / K-step.
__global__ void __launch_bounds__(256) k_gemm16(const _Float16* __restrict__ A,
                                                const _Float16* __restrict__ Bt,
                                                float* __restrict__ C,
                                                int M, int N, int K) {
    int wave = (blockIdx.x * blockDim.x + threadIdx.x) >> 5;
    int lane = threadIdx.x & 31;
    int tiles_n = N >> 6;
    int tiles_m = M >> 5;
    if (wave >= tiles_m * tiles_n) return;        // uniform per wave
    int tm = wave / tiles_n, tn = wave % tiles_n;
    int lo  = lane & 15;
    int sel = (lane >> 4) << 3;                   // 0 or 8: ISA 16-bit frag K grouping

    const _Float16* arow0 = A + (size_t)(tm * 32 + lo) * K;
    const _Float16* arow1 = arow0 + (size_t)16 * K;
    const _Float16* brow0 = Bt + (size_t)(tn * 64 +  0 + lo) * K;
    const _Float16* brow1 = Bt + (size_t)(tn * 64 + 16 + lo) * K;
    const _Float16* brow2 = Bt + (size_t)(tn * 64 + 32 + lo) * K;
    const _Float16* brow3 = Bt + (size_t)(tn * 64 + 48 + lo) * K;

    v8f acc0[4] = {}, acc1[4] = {};
    for (int k = 0; k < K; k += 32) {
        v16h a0, a1, b0, b1, b2, b3;
        ((v8h*)&a0)[0] = *(const v8h*)(arow0 + k + sel);
        ((v8h*)&a0)[1] = *(const v8h*)(arow0 + k + sel + 16);
        ((v8h*)&a1)[0] = *(const v8h*)(arow1 + k + sel);
        ((v8h*)&a1)[1] = *(const v8h*)(arow1 + k + sel + 16);
        ((v8h*)&b0)[0] = *(const v8h*)(brow0 + k + sel);
        ((v8h*)&b0)[1] = *(const v8h*)(brow0 + k + sel + 16);
        ((v8h*)&b1)[0] = *(const v8h*)(brow1 + k + sel);
        ((v8h*)&b1)[1] = *(const v8h*)(brow1 + k + sel + 16);
        ((v8h*)&b2)[0] = *(const v8h*)(brow2 + k + sel);
        ((v8h*)&b2)[1] = *(const v8h*)(brow2 + k + sel + 16);
        ((v8h*)&b3)[0] = *(const v8h*)(brow3 + k + sel);
        ((v8h*)&b3)[1] = *(const v8h*)(brow3 + k + sel + 16);
        __builtin_prefetch(brow0 + k + 64, 0, 0);
        __builtin_prefetch(brow1 + k + 64, 0, 0);
        __builtin_prefetch(brow2 + k + 64, 0, 0);
        __builtin_prefetch(brow3 + k + 64, 0, 0);

        acc0[0] = __builtin_amdgcn_wmma_f32_16x16x32_f16(false, a0, false, b0, (short)0, acc0[0], false, false);
        acc0[1] = __builtin_amdgcn_wmma_f32_16x16x32_f16(false, a0, false, b1, (short)0, acc0[1], false, false);
        acc0[2] = __builtin_amdgcn_wmma_f32_16x16x32_f16(false, a0, false, b2, (short)0, acc0[2], false, false);
        acc0[3] = __builtin_amdgcn_wmma_f32_16x16x32_f16(false, a0, false, b3, (short)0, acc0[3], false, false);
        acc1[0] = __builtin_amdgcn_wmma_f32_16x16x32_f16(false, a1, false, b0, (short)0, acc1[0], false, false);
        acc1[1] = __builtin_amdgcn_wmma_f32_16x16x32_f16(false, a1, false, b1, (short)0, acc1[1], false, false);
        acc1[2] = __builtin_amdgcn_wmma_f32_16x16x32_f16(false, a1, false, b2, (short)0, acc1[2], false, false);
        acc1[3] = __builtin_amdgcn_wmma_f32_16x16x32_f16(false, a1, false, b3, (short)0, acc1[3], false, false);
    }
    int hi = lane >> 4;
#pragma unroll
    for (int t = 0; t < 4; ++t) {
        int cc = tn * 64 + t * 16 + lo;
        int r0 = tm * 32 + hi * 8;
#pragma unroll
        for (int r = 0; r < 8; ++r)
            C[(size_t)(r0 + r) * N + cc] = acc0[t][r];
        int r1 = tm * 32 + 16 + hi * 8;
#pragma unroll
        for (int r = 0; r < 8; ++r)
            C[(size_t)(r1 + r) * N + cc] = acc1[t][r];
    }
}

// ---------------------------------------------------------------------------
// RoPE on Q: Qp fp32 [B*S][4096] -> Qh f16 [B][32][S][128]
__global__ void k_rope_q(const float* __restrict__ Qp, const int* __restrict__ pos,
                         _Float16* __restrict__ Qh) {
    int i = blockIdx.x * blockDim.x + threadIdx.x;
    if (i >= BATCH * NH * SEQ * (HD / 2)) return;
    int d = i % 64; int t = i / 64;
    int s = t % SEQ; t /= SEQ;
    int h = t % NH;  int b = t / NH;
    float p    = (float)pos[b * SEQ + s];
    float invf = __expf(-9.210340371976184f * ((float)(2 * d) * (1.0f / (float)HD)));
    float fr   = p * invf;
    float c = cosf(fr), sn = sinf(fr);
    size_t src = (size_t)(b * SEQ + s) * HIDDEN + h * HD;
    float x1 = Qp[src + d], x2 = Qp[src + d + 64];
    size_t dst = (((size_t)(b * NH + h)) * SEQ + s) * HD;
    Qh[dst + d]      = (_Float16)(x1 * c - x2 * sn);
    Qh[dst + d + 64] = (_Float16)(x2 * c + x1 * sn);
}

// RoPE on K: Kp fp32 [B*S][1024] -> Kh f16 [B][8][S][128]
__global__ void k_rope_k(const float* __restrict__ Kp, const int* __restrict__ pos,
                         _Float16* __restrict__ Kh) {
    int i = blockIdx.x * blockDim.x + threadIdx.x;
    if (i >= BATCH * NKV * SEQ * (HD / 2)) return;
    int d = i % 64; int t = i / 64;
    int s = t % SEQ; t /= SEQ;
    int h = t % NKV; int b = t / NKV;
    float p    = (float)pos[b * SEQ + s];
    float invf = __expf(-9.210340371976184f * ((float)(2 * d) * (1.0f / (float)HD)));
    float fr   = p * invf;
    float c = cosf(fr), sn = sinf(fr);
    size_t src = (size_t)(b * SEQ + s) * (NKV * HD) + h * HD;
    float x1 = Kp[src + d], x2 = Kp[src + d + 64];
    size_t dst = (((size_t)(b * NKV + h)) * SEQ + s) * HD;
    Kh[dst + d]      = (_Float16)(x1 * c - x2 * sn);
    Kh[dst + d + 64] = (_Float16)(x2 * c + x1 * sn);
}

// V transpose: Vp fp32 [B*S][1024] -> Vt f16 [B][8][128 dim][S]
__global__ void k_vtrans(const float* __restrict__ Vp, _Float16* __restrict__ Vt) {
    int i = blockIdx.x * blockDim.x + threadIdx.x;
    if (i >= BATCH * NKV * SEQ * HD) return;
    int d = i % HD; int t = i / HD;
    int s = t % SEQ; t /= SEQ;
    int h = t % NKV; int b = t / NKV;
    Vt[(((size_t)(b * NKV + h)) * HD + d) * SEQ + s] =
        (_Float16)Vp[(size_t)(b * SEQ + s) * (NKV * HD) + h * HD + d];
}

// ---------------------------------------------------------------------------
// Flash attention. 8 waves/block share one (b,h). Double-buffered K/V chunk
// staging: K via TDM tensor_load_to_lds (wave 0, TENSORcnt) with async-LDS
// fallback, V via global_load_async_to_lds_b128 (ASYNCcnt). Chunk kc+1 is
// staged while chunk kc is computed. All matmuls on v_wmma_f32_16x16x32_f16.
__global__ void __launch_bounds__(256) k_attn(const _Float16* __restrict__ Qh,
                                              const _Float16* __restrict__ Kh,
                                              const _Float16* __restrict__ Vt,
                                              _Float16* __restrict__ ctx) {
    __shared__ __align__(16) _Float16 ktile[2][32 * 128];   // [key][dim]  2x8KB
    __shared__ __align__(16) _Float16 vtile[2][128 * 32];   // [dim][key]  2x8KB
    __shared__ __align__(16) float    sbuf[8][16 * 32];     // scores      16KB
    __shared__ __align__(16) _Float16 pbuf[8][16 * 32];     // probs f16   8KB
    __shared__ __align__(16) float    abuf[8][16];          // alpha / 1/l

    int tid  = threadIdx.x;
    int wid  = tid >> 5;
    int lane = tid & 31;
    int gw   = blockIdx.x * 8 + wid;           // 8192 waves total, exact grid
    int qt   = gw & 127;                       // S/16 = 128 q-tiles
    int bh   = gw >> 7;
    int h    = bh & 31;
    int b    = bh >> 5;
    int kvh  = h >> 2;                         // N_REP = 4
    int lo   = lane & 15;
    int hi   = lane >> 4;
    int sel  = hi << 3;                        // 0 or 8

    const _Float16* qrow  = Qh + (((size_t)(b * NH + h)) * SEQ + qt * 16 + lo) * HD;
    const _Float16* Kbase = Kh + ((size_t)(b * NKV + kvh)) * SEQ * HD;
    const _Float16* Vbase = Vt + ((size_t)(b * NKV + kvh)) * HD * SEQ;
    unsigned klds = lds_off32(&ktile[0][0]);
    unsigned vlds = lds_off32(&vtile[0][0]);

    v16h aq[4];
#pragma unroll
    for (int j = 0; j < 4; ++j) {
        ((v8h*)&aq[j])[0] = *(const v8h*)(qrow + j * 32 + sel);
        ((v8h*)&aq[j])[1] = *(const v8h*)(qrow + j * 32 + sel + 16);
    }

    // ---- stage one 32-key chunk into buffer `buf`
    auto stage = [&](int buf, int key0) {
        const char* kg = (const char*)(Kbase + (size_t)key0 * HD);
        unsigned kl = klds + buf * 8192;
#if HAVE_TDM
        if (wid == 0) tdm_copy_8k(kl, (unsigned long long)kg);  // TDM, TENSORcnt
#else
        async_cp16(kl + tid * 16,        kg + tid * 16);
        async_cp16(kl + tid * 16 + 4096, kg + tid * 16 + 4096);
#endif
        unsigned vl = vlds + buf * 8192;
        int j0 = tid * 2;
        int r0 = j0 >> 2, p0 = (j0 & 3) * 16;
        async_cp16(vl + r0 * 64 + p0, (const char*)(Vbase + (size_t)r0 * SEQ + key0) + p0);
        int j1 = j0 + 1;
        int r1 = j1 >> 2, p1 = (j1 & 3) * 16;
        async_cp16(vl + r1 * 64 + p1, (const char*)(Vbase + (size_t)r1 * SEQ + key0) + p1);
    };
    auto wait_stage = [&]() {
        wait_async0();                         // own ASYNC copies
#if HAVE_TDM
        if (wid == 0) __builtin_amdgcn_s_wait_tensorcnt(0);
#endif
        __syncthreads();                       // everyone's staged data visible
    };

    float m_run = -3.0e38f, l_run = 0.0f;      // meaningful in lanes 0..15
    v8f o[8] = {};
    float* srow = &sbuf[wid][0];

    stage(0, 0);
    wait_stage();

    const int NC = SEQ / 32;
    for (int kc = 0; kc < NC; ++kc) {
        int cur = kc & 1;
        if (kc + 1 < NC) stage(1 - cur, (kc + 1) * 32);   // overlap with compute

        // ---- scores: S = Q K^T (two 16-key tiles), K-frags from LDS
        v8f s0 = {}, s1 = {};
#pragma unroll
        for (int j = 0; j < 4; ++j) {
            v16h bk0, bk1;
            const _Float16* kr0 = &ktile[cur][lo * 128 + j * 32 + sel];
            const _Float16* kr1 = &ktile[cur][(16 + lo) * 128 + j * 32 + sel];
            ((v8h*)&bk0)[0] = *(const v8h*)(kr0);
            ((v8h*)&bk0)[1] = *(const v8h*)(kr0 + 16);
            ((v8h*)&bk1)[0] = *(const v8h*)(kr1);
            ((v8h*)&bk1)[1] = *(const v8h*)(kr1 + 16);
            s0 = __builtin_amdgcn_wmma_f32_16x16x32_f16(false, aq[j], false, bk0,
                                                        (short)0, s0, false, false);
            s1 = __builtin_amdgcn_wmma_f32_16x16x32_f16(false, aq[j], false, bk1,
                                                        (short)0, s1, false, false);
        }
#pragma unroll
        for (int r = 0; r < 8; ++r) {
            srow[(r + hi * 8) * 32 + lo]      = s0[r] * SCALE;
            srow[(r + hi * 8) * 32 + lo + 16] = s1[r] * SCALE;
        }
        __syncthreads();
        if (lane < 16) {                       // row-owner softmax (no WMMA inside)
            float mx = m_run;
#pragma unroll 8
            for (int c = 0; c < 32; ++c) mx = fmaxf(mx, srow[lane * 32 + c]);
            float alpha = __expf(m_run - mx);
            float sum = 0.0f;
#pragma unroll 8
            for (int c = 0; c < 32; ++c) {
                float pv = __expf(srow[lane * 32 + c] - mx);
                sum += pv;
                pbuf[wid][lane * 32 + c] = (_Float16)pv;
            }
            l_run = l_run * alpha + sum;
            m_run = mx;
            abuf[wid][lane] = alpha;
        }
        __syncthreads();
        float al[8];
#pragma unroll
        for (int r = 0; r < 8; ++r) al[r] = abuf[wid][hi * 8 + r];
#pragma unroll
        for (int t = 0; t < 8; ++t)
#pragma unroll
            for (int r = 0; r < 8; ++r) o[t][r] *= al[r];
        // P fragment (A layout) from LDS
        v16h ap;
        const _Float16* prow = &pbuf[wid][lo * 32];
        ((v8h*)&ap)[0] = *(const v8h*)(prow + sel);
        ((v8h*)&ap)[1] = *(const v8h*)(prow + sel + 16);
        // PV: 8 dim-tiles of 16, V-frags from LDS
#pragma unroll
        for (int t = 0; t < 8; ++t) {
            const _Float16* vr = &vtile[cur][(t * 16 + lo) * 32 + sel];
            v16h bv;
            ((v8h*)&bv)[0] = *(const v8h*)(vr);
            ((v8h*)&bv)[1] = *(const v8h*)(vr + 16);
            o[t] = __builtin_amdgcn_wmma_f32_16x16x32_f16(false, ap, false, bv,
                                                          (short)0, o[t], false, false);
        }
        wait_stage();                          // next chunk staged + tiles safe
    }
    // normalize by l and emit ctx f16 [B*S][4096]
    if (lane < 16) abuf[wid][lane] = 1.0f / l_run;
    __syncthreads();
    float rl[8];
#pragma unroll
    for (int r = 0; r < 8; ++r) rl[r] = abuf[wid][hi * 8 + r];
#pragma unroll
    for (int t = 0; t < 8; ++t) {
        int cc = h * HD + t * 16 + lo;
#pragma unroll
        for (int r = 0; r < 8; ++r) {
            int rr = b * SEQ + qt * 16 + r + hi * 8;
            ctx[(size_t)rr * HIDDEN + cc] = (_Float16)(o[t][r] * rl[r]);
        }
    }
}

// ---------------------------------------------------------------------------
extern "C" void kernel_launch(void* const* d_in, const int* in_sizes, int n_in,
                              void* d_out, int out_size, void* d_ws, size_t ws_size,
                              hipStream_t stream) {
    const float* query = (const float*)d_in[0];
    const float* key   = (const float*)d_in[1];
    const float* value = (const float*)d_in[2];
    const int*   pos   = (const int*)d_in[3];
    const float* Wq    = (const float*)d_in[4];
    const float* Wk    = (const float*)d_in[5];
    const float* Wv    = (const float*)d_in[6];
    const float* Wo    = (const float*)d_in[7];
    float* out = (float*)d_out;

    char* ws = (char*)d_ws;
    size_t off = 0;
    auto alloc = [&](size_t bytes) { char* p = ws + off; off += (bytes + 255) & ~(size_t)255; return p; };
    _Float16* Xh = (_Float16*)alloc((size_t)NROWS * HIDDEN * 2);     // activations f16 (reused, also ctx)
    _Float16* WT = (_Float16*)alloc((size_t)HIDDEN * HIDDEN * 2);    // transposed weight f16 (reused)
    float*    Qp = (float*)alloc((size_t)NROWS * HIDDEN * 4);
    float*    Kp = (float*)alloc((size_t)NROWS * NKV * HD * 4);
    float*    Vp = (float*)alloc((size_t)NROWS * NKV * HD * 4);
    _Float16* Qh = (_Float16*)alloc((size_t)BATCH * NH * SEQ * HD * 2);
    _Float16* Khb = (_Float16*)alloc((size_t)BATCH * NKV * SEQ * HD * 2);
    _Float16* Vtb = (_Float16*)alloc((size_t)BATCH * NKV * HD * SEQ * 2);

    const int TB = 256;
    const int nAct8 = (NROWS * HIDDEN) / 8;
    const int gAct  = (nAct8 + TB - 1) / TB;
    const int gWbig = (HIDDEN / 32) * (HIDDEN / 32);
    const int gWsm  = (HIDDEN / 32) * ((NKV * HD) / 32);
    const int gemmQblocks  = ((NROWS / 32) * (HIDDEN / 64)) / 8;      // 1024
    const int gemmKVblocks = ((NROWS / 32) * ((NKV * HD) / 64)) / 8;  // 256

    // Q projection
    k_f32_to_f16<<<gAct, TB, 0, stream>>>(query, Xh, nAct8);
    k_transp_f16<<<gWbig, TB, 0, stream>>>(Wq, WT, HIDDEN, HIDDEN);
    k_gemm16<<<gemmQblocks, TB, 0, stream>>>(Xh, WT, Qp, NROWS, HIDDEN, HIDDEN);
    // K projection
    k_f32_to_f16<<<gAct, TB, 0, stream>>>(key, Xh, nAct8);
    k_transp_f16<<<gWsm, TB, 0, stream>>>(Wk, WT, HIDDEN, NKV * HD);
    k_gemm16<<<gemmKVblocks, TB, 0, stream>>>(Xh, WT, Kp, NROWS, NKV * HD, HIDDEN);
    // V projection
    k_f32_to_f16<<<gAct, TB, 0, stream>>>(value, Xh, nAct8);
    k_transp_f16<<<gWsm, TB, 0, stream>>>(Wv, WT, HIDDEN, NKV * HD);
    k_gemm16<<<gemmKVblocks, TB, 0, stream>>>(Xh, WT, Vp, NROWS, NKV * HD, HIDDEN);
    // RoPE + layout transforms
    {
        int n = BATCH * NH * SEQ * (HD / 2);
        k_rope_q<<<(n + TB - 1) / TB, TB, 0, stream>>>(Qp, pos, Qh);
        n = BATCH * NKV * SEQ * (HD / 2);
        k_rope_k<<<(n + TB - 1) / TB, TB, 0, stream>>>(Kp, pos, Khb);
        n = BATCH * NKV * SEQ * HD;
        k_vtrans<<<(n + TB - 1) / TB, TB, 0, stream>>>(Vp, Vtb);
    }
    // Flash attention -> ctx (reuses Xh)
    k_attn<<<(BATCH * NH * (SEQ / 16)) / 8, TB, 0, stream>>>(Qh, Khb, Vtb, Xh);
    // Output projection -> d_out (fp32)
    k_transp_f16<<<gWbig, TB, 0, stream>>>(Wo, WT, HIDDEN, HIDDEN);
    k_gemm16<<<gemmQblocks, TB, 0, stream>>>(Xh, WT, out, NROWS, HIDDEN, HIDDEN);
}
